// TimeUserPair_86406152061244
// MI455X (gfx1250) — matrix operand order repaired and checked by reading, not compiled, not run on hardware
//
#include <hip/hip_runtime.h>

// ---------- types (trivial ext-vectors so they can live in unions) ----------
typedef _Float16     v16h  __attribute__((ext_vector_type(16)));
typedef float        v8f   __attribute__((ext_vector_type(8)));
typedef _Float16     h4    __attribute__((ext_vector_type(4)));
typedef unsigned int u32x4 __attribute__((ext_vector_type(4)));
typedef float        f32x4 __attribute__((ext_vector_type(4)));

union V16 { v16h v; u32x4 q[2]; };

// Problem constants
#define BB 128
#define SS 512
#define TT 24
#define DD 256
#define HH 4
#define HD 64
#define NN (BB*SS)          // 65536
#define WAVES 4

// Workspace layout (in _Float16 elements)
#define WS_WQ   0           // 16nt x 16kt tiles x 512  = 131072
#define WS_UNI  131072      // 16nt x 8kt  tiles x 512  =  65536
#define WS_HW   196608      //  2nt x 8kt  tiles x 512  =   8192
#define WS_KT   204800      //  4h x 2tt x 2kt x 512    =   8192
#define WS_V    212992      //  4h x 4et x 512          =   8192

// Per-wave LDS layout (bytes)
#define L_QRY   0           // 16x512 f16 = 16384
#define L_QB    16384       // 16x256 f16 =  8192
#define L_HD    24576       // 16x256 f16 =  8192
#define L_ZB    32768       // 16x32  f32 =  2048
#define L_PB    34816       // 16x32  f16 =  1024
#define L_WAVE  35840
#define L_TOTAL (L_WAVE*WAVES)

#define WMMA16(a, b, c) __builtin_amdgcn_wmma_f32_16x16x32_f16(false, (a), false, (b), (short)0, (c), false, false)

// ---------------------------------------------------------------------------
// Pre-swizzle Wq / unify_w / head_w into f16 WMMA-B tile layout.
// B-tile element (lane, j): column n = nt*16 + (lane&15),
//                           K = kt*32 + (j>>3)*16 + (lane>>4)*8 + (j&7)
// ---------------------------------------------------------------------------
__global__ void prep_weights_kernel(const float* __restrict__ wq_w,
                                    const float* __restrict__ unify_w,
                                    const float* __restrict__ head_w,
                                    _Float16* __restrict__ ws) {
  int idx = blockIdx.x * blockDim.x + threadIdx.x;
  int e, KT, mode;
  _Float16* dst;
  if (idx < 131072)      { e = idx;          KT = 16; mode = 0; dst = ws + WS_WQ;  }
  else if (idx < 196608) { e = idx - 131072; KT = 8;  mode = 1; dst = ws + WS_UNI; }
  else if (idx < 204800) { e = idx - 196608; KT = 8;  mode = 2; dst = ws + WS_HW;  }
  else return;
  int tile = e >> 9, r = e & 511, lane = r >> 4, j = r & 15;
  int kt = tile % KT,  nt = tile / KT;
  int K = kt * 32 + ((j >> 3) << 4) + ((lane >> 4) << 3) + (j & 7);
  int c = nt * 16 + (lane & 15);
  float v;
  if (mode == 0)      v = wq_w[((size_t)(c >> 6)) * 512 * 64 + (size_t)K * 64 + (c & 63)];
  else if (mode == 1) v = unify_w[(size_t)K * 256 + c];
  else                v = (c < TT) ? head_w[(size_t)K * TT + c] : 0.0f;
  dst[e] = (_Float16)v;
}

// ---------------------------------------------------------------------------
// k = key*Wk + bk, v = key*Wv + bv (tiny), written in swizzled WMMA-B layout:
//   kT : per (h, tt=t/16, kt=e/32) tile, B[K=e%32][n=t%16]
//   v  : per (h, et=e/16)          tile, B[K=t   ][n=e%16], t=24..31 zeroed
// ---------------------------------------------------------------------------
__global__ void prep_kv_kernel(const float* __restrict__ skey,
                               const float* __restrict__ wk_w,
                               const float* __restrict__ wk_b,
                               const float* __restrict__ wv_w,
                               const float* __restrict__ wv_b,
                               _Float16* __restrict__ ws) {
  _Float16* kTP = ws + WS_KT;
  _Float16* vP  = ws + WS_V;
  for (int idx = threadIdx.x; idx < 8192; idx += 256) {
    int h = idx >> 11, rem = idx & 2047, t = rem >> 6, e = rem & 63;
    float kv = 0.f, vv = 0.f;
    if (t < TT) {
      const float* kr  = skey + (size_t)t * DD;
      const float* wkh = wk_w + (size_t)h * DD * HD + e;
      const float* wvh = wv_w + (size_t)h * DD * HD + e;
      for (int d = 0; d < DD; ++d) {
        float x = kr[d];
        kv += x * wkh[(size_t)d * HD];
        vv += x * wvh[(size_t)d * HD];
      }
      kv += wk_b[h * HD + e];
      vv += wv_b[h * HD + e];
    }
    { // kT tile write
      int ttI = t >> 4, kt = e >> 5, Kl = e & 31, nl = t & 15;
      int lane = nl + (((Kl >> 3) & 1) << 4), j = (Kl & 7) + ((Kl >> 4) << 3);
      kTP[((((h * 2 + ttI) * 2) + kt) << 9) + lane * 16 + j] = (_Float16)kv;
    }
    { // v tile write
      int et = e >> 4, Kl = t, nl = e & 15;
      int lane = nl + (((Kl >> 3) & 1) << 4), j = (Kl & 7) + ((Kl >> 4) << 3);
      vP[((h * 4 + et) << 9) + lane * 16 + j] = (_Float16)vv;
    }
  }
}

// ---------------------------------------------------------------------------
// Fully fused forward: one 16-row tile of N per wave.
// ---------------------------------------------------------------------------
__global__ __launch_bounds__(WAVES * 32)
void fused_main_kernel(const float* __restrict__ timeslot,
                       const int*   __restrict__ user,
                       const int*   __restrict__ hour,
                       const int*   __restrict__ hour_mask,
                       const float* __restrict__ user_pref,
                       const float* __restrict__ wq_b,
                       const float* __restrict__ unify_b,
                       const float* __restrict__ head_b,
                       const _Float16* __restrict__ ws,
                       float* __restrict__ at_emb,
                       float* __restrict__ time_logits) {
  extern __shared__ char smem[];
  const _Float16* wqP  = ws + WS_WQ;
  const _Float16* uniP = ws + WS_UNI;
  const _Float16* hwP  = ws + WS_HW;
  const _Float16* kTP  = ws + WS_KT;
  const _Float16* vP   = ws + WS_V;

  const int w = threadIdx.x >> 5;
  const int l = threadIdx.x & 31;
  const int m = l & 15;          // A row / D column index
  const int hi = l >> 4;         // lane half
  char* base = smem + (size_t)w * L_WAVE;
  _Float16* qry = (_Float16*)(base + L_QRY);
  _Float16* qb  = (_Float16*)(base + L_QB);
  _Float16* hd  = (_Float16*)(base + L_HD);
  float*    zb  = (float*)   (base + L_ZB);
  _Float16* pb  = (_Float16*)(base + L_PB);

  const int n0 = (blockIdx.x * WAVES + w) << 4;   // first of 16 rows

  // ---- Step 1: build query[16][512] = [user_pref row | timeslot row] as f16
  for (int r = 0; r < 16; ++r) {
    int n = n0 + r, b = n >> 9, s = n & (SS - 1);
    const float* uf = user_pref + (size_t)user[b] * DD;
    const float* tf = timeslot + (size_t)hour[(b << 9) + s] * DD;
    #pragma unroll
    for (int c = l * 4; c < DD; c += 128) {
      f32x4 f = *(const f32x4*)(uf + c);
      h4 hv; hv.x = (_Float16)f.x; hv.y = (_Float16)f.y; hv.z = (_Float16)f.z; hv.w = (_Float16)f.w;
      *(h4*)(qry + r * 512 + c) = hv;
      f32x4 g = *(const f32x4*)(tf + c);
      h4 hw2; hw2.x = (_Float16)g.x; hw2.y = (_Float16)g.y; hw2.z = (_Float16)g.z; hw2.w = (_Float16)g.w;
      *(h4*)(qry + r * 512 + DD + c) = hw2;
    }
  }

  // ---- Step 2: q = query * Wq + bq  ->  qb[16][256] f16
  for (int nt = 0; nt < 16; ++nt) {
    float bias = wq_b[nt * 16 + m];
    v8f acc = {bias, bias, bias, bias, bias, bias, bias, bias};
    #pragma unroll
    for (int kt = 0; kt < 16; ++kt) {
      V16 a, bmat;
      const char* ap = (const char*)qry + m * 1024 + kt * 64 + hi * 16;
      a.q[0] = *(const u32x4*)ap;
      a.q[1] = *(const u32x4*)(ap + 32);
      const u32x4* bp = (const u32x4*)(wqP + ((nt * 16 + kt) << 9) + l * 16);
      bmat.q[0] = bp[0]; bmat.q[1] = bp[1];
      acc = WMMA16(a.v, bmat.v, acc);
    }
    #pragma unroll
    for (int r2 = 0; r2 < 8; ++r2)
      qb[(r2 + hi * 8) * 256 + nt * 16 + m] = (_Float16)acc[r2];
  }

  // ---- Step 3: per-head attention (T=24, padded to 32)
  const float scale = 0.125f;   // 1/sqrt(64)
  for (int h = 0; h < HH; ++h) {
    // scores -> z
    #pragma unroll
    for (int tt = 0; tt < 2; ++tt) {
      v8f acc = {};
      #pragma unroll
      for (int kt = 0; kt < 2; ++kt) {
        V16 a, bmat;
        const char* ap = (const char*)qb + m * 512 + h * 128 + kt * 64 + hi * 16;
        a.q[0] = *(const u32x4*)ap;
        a.q[1] = *(const u32x4*)(ap + 32);
        const u32x4* bp = (const u32x4*)(kTP + ((((h * 2 + tt) * 2) + kt) << 9) + l * 16);
        bmat.q[0] = bp[0]; bmat.q[1] = bp[1];
        acc = WMMA16(a.v, bmat.v, acc);
      }
      int t = tt * 16 + m;
      if (t < TT) {
        #pragma unroll
        for (int r2 = 0; r2 < 8; ++r2) {
          int mm = r2 + hi * 8;
          float sv = acc[r2] * scale;
          int mk = hour_mask[(size_t)(n0 + mm) * TT + t];
          // oscillator collapses to relu(score); masked (-inf) -> 0
          zb[mm * 32 + t] = (mk == 1) ? 0.f : fmaxf(sv, 0.f);
        }
      }
    }
    // softmax over 24 (z >= 0, masked entries participate with z = 0)
    if (l < 16) {
      float mx = 0.f;
      for (int t = 0; t < TT; ++t) mx = fmaxf(mx, zb[l * 32 + t]);
      float ex[TT], sum = 0.f;
      for (int t = 0; t < TT; ++t) { ex[t] = __expf(zb[l * 32 + t] - mx); sum += ex[t]; }
      float inv = 1.f / sum;
      for (int t = 0; t < TT; ++t) pb[l * 32 + t] = (_Float16)(ex[t] * inv);
      for (int t = TT; t < 32; ++t) pb[l * 32 + t] = (_Float16)0.f;
    }
    // weighted = probs * v  -> heads columns [h*64 .. h*64+63]
    #pragma unroll
    for (int et = 0; et < 4; ++et) {
      v8f acc = {};
      V16 a, bmat;
      const char* ap = (const char*)pb + m * 64 + hi * 16;
      a.q[0] = *(const u32x4*)ap;
      a.q[1] = *(const u32x4*)(ap + 32);
      const u32x4* bp = (const u32x4*)(vP + ((h * 4 + et) << 9) + l * 16);
      bmat.q[0] = bp[0]; bmat.q[1] = bp[1];
      acc = WMMA16(a.v, bmat.v, acc);
      #pragma unroll
      for (int r2 = 0; r2 < 8; ++r2)
        hd[(r2 + hi * 8) * 256 + h * 64 + et * 16 + m] = (_Float16)acc[r2];
    }
  }

  // ---- Step 4a: at_emb = heads * unify_w + unify_b
  for (int nt = 0; nt < 16; ++nt) {
    float bias = unify_b[nt * 16 + m];
    v8f acc = {bias, bias, bias, bias, bias, bias, bias, bias};
    #pragma unroll
    for (int kt = 0; kt < 8; ++kt) {
      V16 a, bmat;
      const char* ap = (const char*)hd + m * 512 + kt * 64 + hi * 16;
      a.q[0] = *(const u32x4*)ap;
      a.q[1] = *(const u32x4*)(ap + 32);
      const u32x4* bp = (const u32x4*)(uniP + ((nt * 8 + kt) << 9) + l * 16);
      bmat.q[0] = bp[0]; bmat.q[1] = bp[1];
      acc = WMMA16(a.v, bmat.v, acc);
    }
    #pragma unroll
    for (int r2 = 0; r2 < 8; ++r2)
      at_emb[(size_t)(n0 + r2 + hi * 8) * 256 + nt * 16 + m] = acc[r2];
  }

  // ---- Step 4b: time_logits = heads * head_w + head_b (cols >= 24 discarded)
  for (int nt = 0; nt < 2; ++nt) {
    int col = nt * 16 + m;
    float bias = (col < TT) ? head_b[col] : 0.f;
    v8f acc = {bias, bias, bias, bias, bias, bias, bias, bias};
    #pragma unroll
    for (int kt = 0; kt < 8; ++kt) {
      V16 a, bmat;
      const char* ap = (const char*)hd + m * 512 + kt * 64 + hi * 16;
      a.q[0] = *(const u32x4*)ap;
      a.q[1] = *(const u32x4*)(ap + 32);
      const u32x4* bp = (const u32x4*)(hwP + ((nt * 8 + kt) << 9) + l * 16);
      bmat.q[0] = bp[0]; bmat.q[1] = bp[1];
      acc = WMMA16(a.v, bmat.v, acc);
    }
    if (col < TT) {
      #pragma unroll
      for (int r2 = 0; r2 < 8; ++r2)
        time_logits[(size_t)(n0 + r2 + hi * 8) * TT + col] = acc[r2];
    }
  }
}

// ---------------------------------------------------------------------------
extern "C" void kernel_launch(void* const* d_in, const int* in_sizes, int n_in,
                              void* d_out, int out_size, void* d_ws, size_t ws_size,
                              hipStream_t stream) {
  const float* timeslot  = (const float*)d_in[0];
  const float* skey      = (const float*)d_in[1];   // smoothed_timeslot_embedded
  const int*   user      = (const int*)  d_in[3];
  const int*   hour      = (const int*)  d_in[4];
  const int*   hour_mask = (const int*)  d_in[5];
  const float* user_pref = (const float*)d_in[6];
  const float* wq_w      = (const float*)d_in[7];
  const float* wq_b      = (const float*)d_in[8];
  const float* wk_w      = (const float*)d_in[9];
  const float* wk_b      = (const float*)d_in[10];
  const float* wv_w      = (const float*)d_in[11];
  const float* wv_b      = (const float*)d_in[12];
  const float* unify_w   = (const float*)d_in[13];
  const float* unify_b   = (const float*)d_in[14];
  const float* head_w    = (const float*)d_in[15];
  const float* head_b    = (const float*)d_in[16];

  _Float16* ws = (_Float16*)d_ws;
  float* at_emb = (float*)d_out;
  float* time_logits = at_emb + (size_t)BB * SS * DD;

  prep_weights_kernel<<<800, 256, 0, stream>>>(wq_w, unify_w, head_w, ws);
  prep_kv_kernel<<<1, 256, 0, stream>>>(skey, wk_w, wk_b, wv_w, wv_b, ws);

  (void)hipFuncSetAttribute((const void*)fused_main_kernel,
                            hipFuncAttributeMaxDynamicSharedMemorySize, L_TOTAL);
  fused_main_kernel<<<NN / 16 / WAVES, WAVES * 32, L_TOTAL, stream>>>(
      timeslot, user, hour, hour_mask, user_pref,
      wq_b, unify_b, head_b, ws, at_emb, time_logits);
}
